// SimpleResRNN_14680198218274
// MI455X (gfx1250) — matrix operand order repaired and checked by reading, not compiled, not run on hardware
//
#include <hip/hip_runtime.h>
#include <math.h>

#define T_STEPS 256
#define BATCH   128
#define IN      512
#define HID     1536
#define OUTN    256
#define KTOT    (IN + HID)      // 2048
#define DT_C    0.01f
#define GAMMA_C 0.9f
#define THETA_C 1.0f

typedef __attribute__((ext_vector_type(16))) __bf16 v16bf;
typedef __attribute__((ext_vector_type(8)))  __bf16 v8bf;
typedef __attribute__((ext_vector_type(8)))  float  v8f;

__device__ __forceinline__ unsigned short f2bf(float f) {
    union { float f; unsigned u; } c; c.f = f;
    unsigned u = c.u;
    u += 0x7FFFu + ((u >> 16) & 1u);      // round-to-nearest-even
    return (unsigned short)(u >> 16);
}
__device__ __forceinline__ float bf2f(unsigned short h) {
    union { unsigned u; float f; } c; c.u = ((unsigned)h) << 16;
    return c.f;
}

// ---------------- preprocessing ----------------
__global__ void k_cvt_bf16(const float* __restrict__ src,
                           unsigned short* __restrict__ dst, int n) {
    int i = blockIdx.x * blockDim.x + threadIdx.x;
    int s = gridDim.x * blockDim.x;
    for (; i < n; i += s) dst[i] = f2bf(src[i]);
}

__global__ void k_init_state(float* __restrict__ u, float* __restrict__ v,
                             float* __restrict__ q, float* __restrict__ ou,
                             unsigned short* __restrict__ z0) {
    int i = blockIdx.x * blockDim.x + threadIdx.x;
    int s = gridDim.x * blockDim.x;
    for (int j = i; j < BATCH * HID; j += s) {
        u[j] = 0.f; v[j] = 0.f; q[j] = 0.f; z0[j] = 0;
    }
    for (int j = i; j < BATCH * OUTN; j += s) ou[j] = 0.f;
}

// ---------------- hidden step ------------------------------------------
// cur = [x_t, z] @ Wh^T (16x64 tile per block), fused oscillator + spike.
// grid = (HID/64, BATCH/16), block = 128 (4 waves). K=2048 split across
// the 4 waves (wave 0: x region K=512; waves 1-3: z region, 512 each);
// partial C tiles reduced through LDS, epilogue coalesced across block.
__global__ void __launch_bounds__(128) k_hidden(
    const unsigned short* __restrict__ xb,    // (T,B,IN)   bf16
    const unsigned short* __restrict__ Wh,    // (HID,KTOT) bf16
    const unsigned short* __restrict__ zprev, // (B,HID)    bf16
    unsigned short* __restrict__ znext,       // (B,HID)    bf16
    const float* __restrict__ omega,
    const float* __restrict__ boff,
    float* __restrict__ u, float* __restrict__ v, float* __restrict__ q,
    int* __restrict__ partials, int nblk, int t)
{
    const int tid  = threadIdx.x;
    const int lane = tid & 31;
    const int wave = tid >> 5;       // 0..3 = K-slice
    const int kh   = lane >> 4;
    const int lm   = lane & 15;
    const int h0   = blockIdx.x * 64;
    const int b0   = blockIdx.y * 16;

    // per-wave K slice: wave 0 -> x[0..512); wave w>0 -> z[(w-1)*512 ..)
    const unsigned short* aptr;
    int bbase;
    if (wave == 0) {
        aptr  = xb + ((size_t)t * BATCH + (b0 + lm)) * IN;
        bbase = 0;
    } else {
        aptr  = zprev + (size_t)(b0 + lm) * HID + (wave - 1) * 512;
        bbase = IN + (wave - 1) * 512;
    }

    v8f acc[4] = {};
    for (int k0 = 0; k0 < 512; k0 += 32) {
        v8bf alo = *(const v8bf*)(aptr + k0 + kh * 16);
        v8bf ahi = *(const v8bf*)(aptr + k0 + kh * 16 + 8);
        v16bf a = __builtin_shufflevector(alo, ahi,
                    0,1,2,3,4,5,6,7,8,9,10,11,12,13,14,15);
#pragma unroll
        for (int j = 0; j < 4; ++j) {
            int n = h0 + lm + 16 * j;
            v16bf b = *(const v16bf*)(Wh + (size_t)n * KTOT + bbase + k0 + kh * 16);
            acc[j] = __builtin_amdgcn_wmma_f32_16x16x32_bf16(
                         false, a, false, b, (short)0, acc[j], false, false);
        }
    }

    // reduce partial tiles through LDS (canonical m*64+n layout)
    __shared__ float ctile[4][16 * 64];
#pragma unroll
    for (int j = 0; j < 4; ++j)
#pragma unroll
        for (int r = 0; r < 8; ++r)
            ctile[wave][(r + 8 * kh) * 64 + lm + 16 * j] = acc[j][r];
    __syncthreads();

    // coalesced fused epilogue: thread owns 8 contiguous columns of one row
    const int m  = tid >> 3;          // 0..15
    const int n0 = (tid & 7) * 8;     // 0,8,..,56
    const int b_row = b0 + m;
    int cnt = 0;
#pragma unroll
    for (int i = 0; i < 8; ++i) {
        int n = n0 + i;
        int h = h0 + n;
        int e = m * 64 + n;
        float cur = ctile[0][e] + ctile[1][e] + ctile[2][e] + ctile[3][e];
        float om = omega[h];
        float po = (-1.f + sqrtf(1.f - (DT_C * om) * (DT_C * om))) / DT_C;
        size_t idx = (size_t)b_row * HID + h;
        float uu = u[idx], vv = v[idx], qq = q[idx];
        float be = po - boff[h] - qq;
        float un = uu + DT_C * (be * uu - om * vv + cur);
        float vn = vv + DT_C * (om * uu + be * vv);
        float z  = (un - THETA_C - qq) > 0.f ? 1.f : 0.f;
        u[idx] = un; v[idx] = vn;
        q[idx] = GAMMA_C * qq + z;
        znext[idx] = f2bf(z);
        cnt += (int)z;
    }

    __shared__ int sred[128];
    sred[tid] = cnt;
    __syncthreads();
    for (int w = 64; w > 0; w >>= 1) {
        if (tid < w) sred[tid] += sred[tid + w];
        __syncthreads();
    }
    if (tid == 0)
        partials[t * nblk + blockIdx.y * gridDim.x + blockIdx.x] = sred[0];
}

// ---------------- output step ------------------------------------------
// out = a*out + (1-a)*(z @ Wo^T). grid=(OUTN/64, BATCH/16), block=128;
// K=1536 split across 4 waves (384 each), LDS reduce, coalesced epilogue.
__global__ void __launch_bounds__(128) k_output(
    const unsigned short* __restrict__ z,   // (B,HID)    bf16
    const unsigned short* __restrict__ Wo,  // (OUTN,HID) bf16
    const float* __restrict__ tau,
    float* __restrict__ out_u,              // (B,OUTN) state
    float* __restrict__ outs,               // (T,B,OUTN) dest
    int t)
{
    const int tid  = threadIdx.x;
    const int lane = tid & 31;
    const int wave = tid >> 5;
    const int kh   = lane >> 4;
    const int lm   = lane & 15;
    const int o0   = blockIdx.x * 64;
    const int b0   = blockIdx.y * 16;

    const int kbase = wave * (HID / 4);   // 384 per wave
    const unsigned short* arow = z + (size_t)(b0 + lm) * HID + kbase;

    v8f acc[4] = {};
    for (int k0 = 0; k0 < HID / 4; k0 += 32) {
        v8bf alo = *(const v8bf*)(arow + k0 + kh * 16);
        v8bf ahi = *(const v8bf*)(arow + k0 + kh * 16 + 8);
        v16bf a = __builtin_shufflevector(alo, ahi,
                    0,1,2,3,4,5,6,7,8,9,10,11,12,13,14,15);
#pragma unroll
        for (int j = 0; j < 4; ++j) {
            int n = o0 + lm + 16 * j;
            v16bf b = *(const v16bf*)(Wo + (size_t)n * HID + kbase + k0 + kh * 16);
            acc[j] = __builtin_amdgcn_wmma_f32_16x16x32_bf16(
                         false, a, false, b, (short)0, acc[j], false, false);
        }
    }

    __shared__ float ctile[4][16 * 64];
#pragma unroll
    for (int j = 0; j < 4; ++j)
#pragma unroll
        for (int r = 0; r < 8; ++r)
            ctile[wave][(r + 8 * kh) * 64 + lm + 16 * j] = acc[j][r];
    __syncthreads();

    const int m  = tid >> 3;
    const int n0 = (tid & 7) * 8;
    const int b_row = b0 + m;
#pragma unroll
    for (int i = 0; i < 8; ++i) {
        int n = n0 + i;
        int o = o0 + n;
        int e = m * 64 + n;
        float d = ctile[0][e] + ctile[1][e] + ctile[2][e] + ctile[3][e];
        float al = __expf(-DT_C / tau[o]);
        size_t idx = (size_t)b_row * OUTN + o;
        float nv = al * out_u[idx] + (1.f - al) * d;
        out_u[idx] = nv;
        outs[(size_t)t * BATCH * OUTN + idx] = nv;
    }
}

// ---------------- finalize: zf, uf, of --------------------------------
__global__ void k_finalize(const unsigned short* __restrict__ zfin,
                           const float* __restrict__ u,
                           const float* __restrict__ out_u,
                           float* __restrict__ d_out) {
    float* zf = d_out + (size_t)T_STEPS * BATCH * OUTN;
    float* uf = zf + (size_t)BATCH * HID;
    float* of = uf + (size_t)BATCH * HID;
    int i = blockIdx.x * blockDim.x + threadIdx.x;
    int s = gridDim.x * blockDim.x;
    for (int j = i; j < BATCH * HID; j += s) {
        zf[j] = bf2f(zfin[j]);
        uf[j] = u[j];
    }
    for (int j = i; j < BATCH * OUTN; j += s) of[j] = out_u[j];
}

__global__ void k_spikesum(const int* __restrict__ partials, int n,
                           float* __restrict__ dst) {
    __shared__ int s[256];
    int tid = threadIdx.x;
    int acc = 0;
    for (int i = tid; i < n; i += 256) acc += partials[i];
    s[tid] = acc;
    __syncthreads();
    for (int w = 128; w > 0; w >>= 1) {
        if (tid < w) s[tid] += s[tid + w];
        __syncthreads();
    }
    if (tid == 0) *dst = (float)s[0];
}

// ---------------- host side -------------------------------------------
extern "C" void kernel_launch(void* const* d_in, const int* in_sizes, int n_in,
                              void* d_out, int out_size, void* d_ws, size_t ws_size,
                              hipStream_t stream) {
    const float* x     = (const float*)d_in[0];
    const float* Wh_f  = (const float*)d_in[1];
    const float* omega = (const float*)d_in[2];
    const float* boff  = (const float*)d_in[3];
    const float* Wo_f  = (const float*)d_in[4];
    const float* tau   = (const float*)d_in[5];
    float* out = (float*)d_out;

    char* ws = (char*)d_ws;
    size_t off = 0;
    auto carve = [&](size_t bytes) -> void* {
        void* p = ws + off;
        off += (bytes + 255) & ~(size_t)255;
        return p;
    };
    unsigned short* xb  = (unsigned short*)carve((size_t)T_STEPS * BATCH * IN * 2);
    unsigned short* Whb = (unsigned short*)carve((size_t)HID * KTOT * 2);
    unsigned short* Wob = (unsigned short*)carve((size_t)OUTN * HID * 2);
    unsigned short* z0  = (unsigned short*)carve((size_t)BATCH * HID * 2);
    unsigned short* z1  = (unsigned short*)carve((size_t)BATCH * HID * 2);
    float* u  = (float*)carve((size_t)BATCH * HID * 4);
    float* v  = (float*)carve((size_t)BATCH * HID * 4);
    float* q  = (float*)carve((size_t)BATCH * HID * 4);
    float* ou = (float*)carve((size_t)BATCH * OUTN * 4);
    const int nblkA = (HID / 64) * (BATCH / 16);   // 192
    int* partials = (int*)carve((size_t)T_STEPS * nblkA * 4);

    k_cvt_bf16<<<1024, 256, 0, stream>>>(x, xb, T_STEPS * BATCH * IN);
    k_cvt_bf16<<<512, 256, 0, stream>>>(Wh_f, Whb, HID * KTOT);
    k_cvt_bf16<<<128, 256, 0, stream>>>(Wo_f, Wob, OUTN * HID);
    k_init_state<<<512, 256, 0, stream>>>(u, v, q, ou, z0);

    for (int t = 0; t < T_STEPS; ++t) {
        unsigned short* zp = (t & 1) ? z1 : z0;
        unsigned short* zn = (t & 1) ? z0 : z1;
        k_hidden<<<dim3(HID / 64, BATCH / 16), 128, 0, stream>>>(
            xb, Whb, zp, zn, omega, boff, u, v, q, partials, nblkA, t);
        k_output<<<dim3(OUTN / 64, BATCH / 16), 128, 0, stream>>>(
            zn, Wob, tau, ou, out, t);
    }
    unsigned short* zfin = ((T_STEPS - 1) & 1) ? z0 : z1;  // T even -> z0

    k_finalize<<<512, 256, 0, stream>>>(zfin, u, ou, out);
    size_t ss_off = (size_t)T_STEPS * BATCH * OUTN
                  + 2 * (size_t)BATCH * HID + (size_t)BATCH * OUTN;
    k_spikesum<<<1, 256, 0, stream>>>(partials, T_STEPS * nblkA, out + ss_off);
}
